// TinyBlock3_34325378629757
// MI455X (gfx1250) — compile-verified, matrix operand order checked by reading, not run on hardware
//
#include <hip/hip_runtime.h>

typedef __attribute__((ext_vector_type(16))) __bf16 v16bf;
typedef __attribute__((ext_vector_type(8)))  float  v8f;

#define T_TOTAL 512
#define NCH     128
#define NPIX    784                      // 28*28
#define OUT_MAIN (T_TOTAL * NCH * NPIX)  // 51380224

// workspace layout (bytes)
#define XBF_BYTES ((size_t)T_TOTAL * NPIX * NCH * 2)   // 102,760,448
#define YBF_OFF   XBF_BYTES
#define WSP_OFF   (2 * XBF_BYTES)
#define WT_OFF    (WSP_OFF + (size_t)NCH * 9 * NCH * 2)
#define ZP_OFF    (WT_OFF + (size_t)NCH * 3 * NCH * 2)   // 1KB zero page

__device__ __forceinline__ unsigned short f2bf(float f) {
    unsigned int u = __float_as_uint(f);
    u += 0x7FFFu + ((u >> 16) & 1u);     // round to nearest even
    return (unsigned short)(u >> 16);
}

union Frag { uint4 u[2]; v16bf v; };

// Build a 16x32 (or 32x16) bf16 WMMA fragment for this lane.
// base points at the lane's K-run start; runs at +0 and +16 bf16 elements.
__device__ __forceinline__ v16bf load_frag(const unsigned short* base) {
    Frag f;
    f.u[0] = *(const uint4*)(base);
    f.u[1] = *(const uint4*)(base + 16);
    return f.v;
}

// ---------------------------------------------------------------------------
// Kernel 0a: transpose-pack x (NCHW f32) -> x_bf [t][pixel][c] bf16
// tile: 128 channels x 112 pixels, via LDS (pad to stride 113: conflict-free)
// ---------------------------------------------------------------------------
__global__ void __launch_bounds__(256) pack_x_kernel(
    const float* __restrict__ x, unsigned short* __restrict__ xbf) {
    __shared__ float tile[NCH][113];
    const int t  = blockIdx.x / 7;
    const int pb = (blockIdx.x % 7) * 112;

    for (int idx = threadIdx.x; idx < NCH * 112; idx += 256) {
        int c = idx / 112, pp = idx % 112;
        tile[c][pp] = x[((size_t)t * NCH + c) * NPIX + pb + pp];
    }
    __syncthreads();

    unsigned int* out = (unsigned int*)xbf;
    for (int idx = threadIdx.x; idx < 64 * 112; idx += 256) {
        int cp = idx & 63, pp = idx >> 6;          // cp = channel pair
        unsigned int v = (unsigned int)f2bf(tile[2 * cp][pp]) |
                         ((unsigned int)f2bf(tile[2 * cp + 1][pp]) << 16);
        out[((size_t)t * NPIX + pb + pp) * 64 + cp] = v;
    }
}

// ---------------------------------------------------------------------------
// Kernel 0b: pack weights to [o][tap][c] bf16; zero the zero-page;
//            copy vid_lens to output tail
// ---------------------------------------------------------------------------
__global__ void __launch_bounds__(256) pack_w_kernel(
    const float* __restrict__ wsp, const float* __restrict__ wt,
    const int* __restrict__ vl,
    unsigned short* __restrict__ wsp_p, unsigned short* __restrict__ wt_p,
    uint4* __restrict__ zpage, int* __restrict__ out_tail) {
    int idx = blockIdx.x * 256 + threadIdx.x;
    const int NSP = NCH * 9 * NCH;
    const int NT  = NCH * 3 * NCH;
    if (idx < NSP) {
        int o = idx / (9 * NCH), r = idx % (9 * NCH);
        int tap = r / NCH, c = r % NCH;
        wsp_p[idx] = f2bf(wsp[(o * NCH + c) * 9 + tap]);   // w_sp[o][c][kh][kw]
    } else if (idx < NSP + NT) {
        int j = idx - NSP;
        int o = j / (3 * NCH), r = j % (3 * NCH);
        int tap = r / NCH, c = r % NCH;
        wt_p[j] = f2bf(wt[(o * NCH + c) * 3 + tap]);       // w_t[o][c][tap]
    }
    if (blockIdx.x == 0) {
        if (threadIdx.x < 64)
            zpage[threadIdx.x] = make_uint4(0u, 0u, 0u, 0u);   // 1KB zeros
        if (threadIdx.x < 8)
            out_tail[threadIdx.x] = vl[threadIdx.x];
    }
}

// ---------------------------------------------------------------------------
// Kernel 1: spatial 3x3 conv, implicit GEMM via v_wmma_f32_16x16x32_bf16
// block = 8 waves; wave w -> out-channels [16w,16w+16); 32 pixels per block.
// Boundary handling: per-tap per-lane pointer select to a zeroed page, so the
// inner K-loop has no EXEC divergence -- just b128 loads + WMMA.
// ---------------------------------------------------------------------------
__global__ void __launch_bounds__(256, 2) conv_sp_kernel(
    const unsigned short* __restrict__ xbf, const unsigned short* __restrict__ wsp,
    const float* __restrict__ bsp, const unsigned short* __restrict__ zp,
    unsigned short* __restrict__ ybf) {
    const int t    = blockIdx.x / 25;
    const int pb   = (blockIdx.x % 25) * 32;
    const int lane = threadIdx.x & 31;
    const int wav  = threadIdx.x >> 5;
    const int mrow = lane & 15;
    const int h2   = lane >> 4;
    const int Mb   = wav * 16;

    const int p0 = pb + mrow,      p1 = pb + 16 + mrow;
    const int h0 = p0 / 28, w0 = p0 % 28;
    const int h1 = p1 / 28, w1 = p1 % 28;

    v8f acc0 = {}, acc1 = {};
    const unsigned short* wbase = wsp + (size_t)(Mb + mrow) * 9 * NCH + h2 * 8;
    const unsigned short* xb    = xbf + (size_t)t * NPIX * NCH + h2 * 8;

    for (int tap = 0; tap < 9; ++tap) {
        const int dh = tap / 3 - 1, dw = tap % 3 - 1;
        const int hh0 = h0 + dh, ww0 = w0 + dw;
        const int hh1 = h1 + dh, ww1 = w1 + dw;
        const bool v0 = (p0 < NPIX) && ((unsigned)hh0 < 28u) && ((unsigned)ww0 < 28u);
        const bool v1 = (p1 < NPIX) && ((unsigned)hh1 < 28u) && ((unsigned)ww1 < 28u);
        const unsigned short* s0 = v0 ? (xb + (ptrdiff_t)(hh0 * 28 + ww0) * NCH) : zp;
        const unsigned short* s1 = v1 ? (xb + (ptrdiff_t)(hh1 * 28 + ww1) * NCH) : zp;
        const unsigned short* wb = wbase + tap * NCH;
#pragma unroll
        for (int cc = 0; cc < 4; ++cc) {
            v16bf a  = load_frag(wb + cc * 32);
            v16bf b0 = load_frag(s0 + cc * 32);
            v16bf b1 = load_frag(s1 + cc * 32);
            acc0 = __builtin_amdgcn_wmma_f32_16x16x32_bf16(
                false, a, false, b0, (short)0, acc0, false, false);
            acc1 = __builtin_amdgcn_wmma_f32_16x16x32_bf16(
                false, a, false, b1, (short)0, acc1, false, false);
        }
    }

    // epilogue: + b_sp, convert to bf16, store y[t][p][c]
    const int cb = Mb + h2 * 8;
    const float4 bl = *(const float4*)(bsp + cb);
    const float4 bh = *(const float4*)(bsp + cb + 4);
    if (p0 < NPIX) {
        uint4 pk;
        pk.x = (unsigned)f2bf(acc0[0] + bl.x) | ((unsigned)f2bf(acc0[1] + bl.y) << 16);
        pk.y = (unsigned)f2bf(acc0[2] + bl.z) | ((unsigned)f2bf(acc0[3] + bl.w) << 16);
        pk.z = (unsigned)f2bf(acc0[4] + bh.x) | ((unsigned)f2bf(acc0[5] + bh.y) << 16);
        pk.w = (unsigned)f2bf(acc0[6] + bh.z) | ((unsigned)f2bf(acc0[7] + bh.w) << 16);
        *(uint4*)(ybf + ((size_t)t * NPIX + p0) * NCH + cb) = pk;
    }
    if (p1 < NPIX) {
        uint4 pk;
        pk.x = (unsigned)f2bf(acc1[0] + bl.x) | ((unsigned)f2bf(acc1[1] + bl.y) << 16);
        pk.y = (unsigned)f2bf(acc1[2] + bl.z) | ((unsigned)f2bf(acc1[3] + bl.w) << 16);
        pk.z = (unsigned)f2bf(acc1[4] + bh.x) | ((unsigned)f2bf(acc1[5] + bh.y) << 16);
        pk.w = (unsigned)f2bf(acc1[6] + bh.z) | ((unsigned)f2bf(acc1[7] + bh.w) << 16);
        *(uint4*)(ybf + ((size_t)t * NPIX + p1) * NCH + cb) = pk;
    }
}

// ---------------------------------------------------------------------------
// Kernel 2: ragged temporal conv (k=3) GEMM + bias + PReLU + residual
// ---------------------------------------------------------------------------
__global__ void __launch_bounds__(256, 2) conv_t_kernel(
    const unsigned short* __restrict__ ybf, const unsigned short* __restrict__ wtp,
    const float* __restrict__ bt, const float* __restrict__ alpha,
    const float* __restrict__ x, const int* __restrict__ vl,
    const unsigned short* __restrict__ zp, float* __restrict__ out) {
    const int t    = blockIdx.x / 25;
    const int pb   = (blockIdx.x % 25) * 32;
    const int lane = threadIdx.x & 31;
    const int wav  = threadIdx.x >> 5;
    const int mrow = lane & 15;
    const int h2   = lane >> 4;
    const int Mb   = wav * 16;

    // segment validity of t-1 / t+1 (uniform per block)
    int prev_ok = 0, next_ok = 0, s = 0;
    for (int i = 0; i < 8; ++i) {
        int e = s + vl[i];
        if (t >= s && t < e) { prev_ok = (t > s); next_ok = (t + 1 < e); }
        s = e;
    }

    const int p0 = pb + mrow, p1 = pb + 16 + mrow;
    v8f acc0 = {}, acc1 = {};

    for (int tap = 0; tap < 3; ++tap) {
        const int dt = tap - 1;
        const bool tvalid = (dt == 0) || (dt < 0 ? (prev_ok != 0) : (next_ok != 0));
        const unsigned short* yb = ybf + (size_t)(t + dt) * NPIX * NCH + h2 * 8;
        const unsigned short* wb = wtp + (size_t)((Mb + mrow) * 3 + tap) * NCH + h2 * 8;
        const bool v0 = tvalid && (p0 < NPIX);
        const bool v1 = tvalid && (p1 < NPIX);
        const unsigned short* s0 = v0 ? (yb + (size_t)p0 * NCH) : zp;
        const unsigned short* s1 = v1 ? (yb + (size_t)p1 * NCH) : zp;
#pragma unroll
        for (int cc = 0; cc < 4; ++cc) {
            v16bf a  = load_frag(wb + cc * 32);
            v16bf b0 = load_frag(s0 + cc * 32);
            v16bf b1 = load_frag(s1 + cc * 32);
            acc0 = __builtin_amdgcn_wmma_f32_16x16x32_bf16(
                false, a, false, b0, (short)0, acc0, false, false);
            acc1 = __builtin_amdgcn_wmma_f32_16x16x32_bf16(
                false, a, false, b1, (short)0, acc1, false, false);
        }
    }

    // epilogue: + b_t, PReLU(alpha), + residual x, store fp32 NCHW
    const int cb = Mb + h2 * 8;
    const float4 bl = *(const float4*)(bt + cb);
    const float4 bh = *(const float4*)(bt + cb + 4);
    const float4 al = *(const float4*)(alpha + cb);
    const float4 ah = *(const float4*)(alpha + cb + 4);
    float bb[8] = { bl.x, bl.y, bl.z, bl.w, bh.x, bh.y, bh.z, bh.w };
    float aa[8] = { al.x, al.y, al.z, al.w, ah.x, ah.y, ah.z, ah.w };

    if (p0 < NPIX) {
#pragma unroll
        for (int v = 0; v < 8; ++v) {
            float z = acc0[v] + bb[v];
            z = (z >= 0.f) ? z : aa[v] * z;
            size_t o = ((size_t)t * NCH + cb + v) * NPIX + p0;
            out[o] = z + x[o];
        }
    }
    if (p1 < NPIX) {
#pragma unroll
        for (int v = 0; v < 8; ++v) {
            float z = acc1[v] + bb[v];
            z = (z >= 0.f) ? z : aa[v] * z;
            size_t o = ((size_t)t * NCH + cb + v) * NPIX + p1;
            out[o] = z + x[o];
        }
    }
}

// ---------------------------------------------------------------------------
extern "C" void kernel_launch(void* const* d_in, const int* in_sizes, int n_in,
                              void* d_out, int out_size, void* d_ws, size_t ws_size,
                              hipStream_t stream) {
    const float* x     = (const float*)d_in[0];
    const int*   vl    = (const int*)d_in[1];
    const float* wsp   = (const float*)d_in[2];
    const float* bsp   = (const float*)d_in[3];
    const float* wt    = (const float*)d_in[4];
    const float* bt    = (const float*)d_in[5];
    const float* alpha = (const float*)d_in[6];

    char* ws = (char*)d_ws;
    unsigned short* xbf  = (unsigned short*)(ws);
    unsigned short* ybf  = (unsigned short*)(ws + YBF_OFF);
    unsigned short* wspp = (unsigned short*)(ws + WSP_OFF);
    unsigned short* wtp  = (unsigned short*)(ws + WT_OFF);
    unsigned short* zp   = (unsigned short*)(ws + ZP_OFF);
    float* out = (float*)d_out;

    pack_x_kernel<<<T_TOTAL * 7, 256, 0, stream>>>(x, xbf);
    pack_w_kernel<<<768, 256, 0, stream>>>(wsp, wt, vl, wspp, wtp,
                                           (uint4*)zp, (int*)d_out + OUT_MAIN);
    conv_sp_kernel<<<T_TOTAL * 25, 256, 0, stream>>>(xbf, wspp, bsp, zp, ybf);
    conv_t_kernel<<<T_TOTAL * 25, 256, 0, stream>>>(ybf, wtp, bt, alpha, x, vl, zp, out);
}
